// HungarianMatcher_86973087744537
// MI455X (gfx1250) — matrix operand order repaired and checked by reading, not compiled, not run on hardware
//
#include <hip/hip_runtime.h>
#include <hip/hip_fp16.h>

// ---------------------------------------------------------------------------
// DETR HungarianMatcher cost matrix for MI455X (gfx1250, wave32).
//   C[b,q,n] = -P[bq, id[n]] + 5*L1(box_bq, tbox_n) - 2*GIoU(...)
// Class gather done as one-hot GEMM on v_wmma_f32_16x16x32_f16 (K=96, f32 acc).
// Target/logit data staged to LDS via the Tensor Data Mover.
// ---------------------------------------------------------------------------

typedef _Float16     v16h  __attribute__((ext_vector_type(16)));
typedef float        v8f   __attribute__((ext_vector_type(8)));
typedef unsigned int u32x4 __attribute__((ext_vector_type(4)));
typedef unsigned int u32x8 __attribute__((ext_vector_type(8)));
typedef int          i32x8 __attribute__((ext_vector_type(8)));
typedef int          i32x4 __attribute__((ext_vector_type(4)));

static constexpr int kB   = 16;
static constexpr int kQ   = 900;
static constexpr int kBQ  = kB * kQ;    // 14400 rows
static constexpr int kNC  = 91;         // classes
static constexpr int kNCP = 96;         // padded K for WMMA (3 x K=32)
static constexpr int kN   = 1600;       // targets
static constexpr int kQT  = 16;         // query rows per block
static constexpr float kBIG = 10000.0f;

__device__ __forceinline__ float clamp01(float x) {
    return fminf(fmaxf(x, 0.0f), 1.0f);
}

// 1-D contiguous TDM copy: elems 32-bit words, global -> LDS.
// D# layout per CDNA5 ISA (group0/group1, groups 2/3 zero).
// This toolchain exposes the 6-arg builtin:
//   (uint32x4 g0, int32x8 g1, int32x4 g2, int32x4 g3, int32x8 extra, i32 cpol)
__device__ __forceinline__ void tdm_load_1d(void* ldsDst, const void* gsrc,
                                            unsigned elems) {
    unsigned lds_off = (unsigned)(unsigned long long)ldsDst;  // low 32b of generic = LDS offset
    unsigned long long ga = (unsigned long long)gsrc;
    u32x4 g0;
    g0.x = 1u;                                                // count=1 (valid user D#)
    g0.y = lds_off;                                           // lds_addr
    g0.z = (unsigned)ga;                                      // global_addr[31:0]
    g0.w = (unsigned)((ga >> 32) & 0x01FFFFFFull) | (2u << 30); // addr[56:32] | type=2
    i32x8 g1;
    g1[0] = (2 << 16);                                   // data_size = 4B
    g1[1] = (int)((elems & 0xFFFFu) << 16);              // tensor_dim0[15:0]
    g1[2] = (int)((elems >> 16) & 0xFFFFu) | (1 << 16);  // tensor_dim0[31:16] | tensor_dim1=1
    g1[3] = (int)((elems & 0xFFFFu) << 16);              // tile_dim0
    g1[4] = 1;                                           // tile_dim1=1, tile_dim2=0
    g1[5] = (int)elems;                                  // tensor_dim0_stride[31:0]
    g1[6] = 0;
    g1[7] = 0;
    i32x4 z4 = {0, 0, 0, 0};
    i32x8 z8 = {0, 0, 0, 0, 0, 0, 0, 0};
    __builtin_amdgcn_tensor_load_to_lds(g0, g1, z4, z4, z8, 0);
}

__global__ __launch_bounds__(256) void matcher_cost_kernel(
    const float* __restrict__ pred_logits,   // [BQ, 91]
    const float* __restrict__ pred_boxes,    // [BQ, 4]
    const int*   __restrict__ tgt_ids,       // [N]
    const float* __restrict__ tgt_boxes,     // [N, 4]
    float*       __restrict__ out)           // [BQ, N]
{
    __shared__ __align__(16) float    sLogit[kQT * kNC];   // staged raw logits
    __shared__ __align__(16) _Float16 sProb[kQT][kNCP];    // f16 softmax probs, zero-padded
    __shared__ __align__(16) float    sQ[kQT][12];         // cx,cy,w,h, x0,y0,x1,y1, area
    __shared__ __align__(16) float    sTB[kN * 4];         // staged target boxes
    __shared__ __align__(16) int      sID[kN];             // staged target ids

    const int tid  = threadIdx.x;
    const int lane = tid & 31;
    const int wv   = tid >> 5;            // 8 waves
    const int r0   = blockIdx.x * kQT;    // first flattened query row of this band

    // ---- Stage inputs with the Tensor Data Mover (wave 0 only) -------------
    if (wv == 0) {
        tdm_load_1d(sLogit, pred_logits + (size_t)r0 * kNC, kQT * kNC);
        tdm_load_1d(sTB,    tgt_boxes,  kN * 4);
        tdm_load_1d(sID,    tgt_ids,    kN);
        __builtin_amdgcn_s_wait_tensorcnt(0);
    }
    // ---- Per-query box precompute (wave 7, lanes 0..15) --------------------
    if (wv == 7 && lane < kQT) {
        const float4 b4 = ((const float4*)pred_boxes)[r0 + lane];
        sQ[lane][0] = b4.x; sQ[lane][1] = b4.y; sQ[lane][2] = b4.z; sQ[lane][3] = b4.w;
        const float cx = clamp01(b4.x), cy = clamp01(b4.y);
        const float w  = clamp01(b4.z), h  = clamp01(b4.w);
        sQ[lane][4] = cx - 0.5f * w;  sQ[lane][5] = cy - 0.5f * h;
        sQ[lane][6] = cx + 0.5f * w;  sQ[lane][7] = cy + 0.5f * h;
        sQ[lane][8] = w * h;
    }
    __syncthreads();

    // ---- Softmax: each wave owns 2 rows; 3 classes per lane ---------------
    #pragma unroll
    for (int i = 0; i < 2; ++i) {
        const int rl = wv * 2 + i;
        const float v0 = sLogit[rl * kNC + lane];                 // lane < 91 always
        const float v1 = sLogit[rl * kNC + lane + 32];            // lane+32 < 91 always
        const float v2 = (lane + 64 < kNC) ? sLogit[rl * kNC + lane + 64] : -1e30f;
        float mx = fmaxf(v0, fmaxf(v1, v2));
        #pragma unroll
        for (int s = 16; s > 0; s >>= 1) mx = fmaxf(mx, __shfl_xor(mx, s, 32));
        const float e0 = __expf(v0 - mx);
        const float e1 = __expf(v1 - mx);
        const float e2 = (lane + 64 < kNC) ? __expf(v2 - mx) : 0.0f;
        float sum = e0 + e1 + e2;
        #pragma unroll
        for (int s = 16; s > 0; s >>= 1) sum += __shfl_xor(sum, s, 32);
        const float inv = __builtin_amdgcn_rcpf(sum);
        sProb[rl][lane]      = (_Float16)(e0 * inv);
        sProb[rl][lane + 32] = (_Float16)(e1 * inv);
        sProb[rl][lane + 64] = (_Float16)(e2 * inv);   // classes 91..95 -> 0
    }
    __syncthreads();

    // ---- A fragments (probs), fixed for all target tiles of this band -----
    // A k order used here (contiguous, not AMD's table order): element e of lane holds K = e + 16*half
    const int row  = lane & 15;
    const int half = lane >> 4;
    v16h a0, a1, a2;
    #pragma unroll
    for (int e = 0; e < 16; ++e) {
        const int k = e + 16 * half;
        a0[e] = sProb[row][k];
        a1[e] = sProb[row][32 + k];
        a2[e] = sProb[row][64 + k];
    }

    // ---- Sweep 16-target tiles; 3 WMMAs/tile for the class-cost GEMM ------
    for (int t = wv; t < kN / 16; t += 8) {
        const int n0 = t * 16;
        const int id = sID[n0 + row];                       // this lane's target column
        const float4 tb = ((const float4*)sTB)[n0 + row];   // raw cxcywh
        const float tcx = clamp01(tb.x), tcy = clamp01(tb.y);
        const float tw  = clamp01(tb.z), th  = clamp01(tb.w);
        const float tx0 = tcx - 0.5f * tw, ty0 = tcy - 0.5f * th;
        const float tx1 = tcx + 0.5f * tw, ty1 = tcy + 0.5f * th;
        const float ta  = tw * th;

        // One-hot B fragments. same contiguous k order as A (not AMD's table order): element e holds K = e + 16*half,
        // i.e. fragment f has its '1' at element (id - 16*half - 32*f) when in [0,16).
        // Build at 32-bit word granularity: one cmp+cndmask per B VGPR (24 total)
        // instead of per f16 element (48). Unsigned wraparound makes out-of-range
        // selectors compare false for every word.
        const unsigned base = (unsigned)(id - 16 * half);
        const unsigned lo   = 0x3C00u << ((base & 1u) << 4);  // f16 1.0 in the right half-word
        const unsigned h0   = base >> 1;                      // word index within fragment 0
        u32x8 w0, w1, w2;
        #pragma unroll
        for (int i = 0; i < 8; ++i) {
            w0[i] = (h0 == (unsigned)i)        ? lo : 0u;
            w1[i] = (h0 == (unsigned)(i + 16)) ? lo : 0u;
            w2[i] = (h0 == (unsigned)(i + 32)) ? lo : 0u;
        }
        const v16h b0 = __builtin_bit_cast(v16h, w0);
        const v16h b1 = __builtin_bit_cast(v16h, w1);
        const v16h b2 = __builtin_bit_cast(v16h, w2);

        v8f acc = {0.f, 0.f, 0.f, 0.f, 0.f, 0.f, 0.f, 0.f};
        acc = __builtin_amdgcn_wmma_f32_16x16x32_f16(false, a0, false, b0,
                                                     (short)0, acc, false, false);
        acc = __builtin_amdgcn_wmma_f32_16x16x32_f16(false, a1, false, b1,
                                                     (short)0, acc, false, false);
        acc = __builtin_amdgcn_wmma_f32_16x16x32_f16(false, a2, false, b2,
                                                     (short)0, acc, false, false);

        // C/D layout: VGPR r -> M = r + 8*half, N = lane&15
        #pragma unroll
        for (int r = 0; r < 8; ++r) {
            const int m = r + 8 * half;
            const float qcx = sQ[m][0], qcy = sQ[m][1], qw = sQ[m][2], qh = sQ[m][3];
            const float qx0 = sQ[m][4], qy0 = sQ[m][5];
            const float qx1 = sQ[m][6], qy1 = sQ[m][7], qa = sQ[m][8];

            // L1 on raw cxcywh
            const float l1 = fabsf(qcx - tb.x) + fabsf(qcy - tb.y) +
                             fabsf(qw  - tb.z) + fabsf(qh  - tb.w);
            // intersection / union
            const float ix0 = fmaxf(qx0, tx0), iy0 = fmaxf(qy0, ty0);
            const float ix1 = fminf(qx1, tx1), iy1 = fminf(qy1, ty1);
            const float iw  = fmaxf(ix1 - ix0, 0.0f);
            const float ih  = fmaxf(iy1 - iy0, 0.0f);
            const float inter = iw * ih;
            const float uni   = qa + ta - inter;
            // enclosing box
            const float cx0 = fminf(qx0, tx0), cy0 = fminf(qy0, ty0);
            const float cx1 = fmaxf(qx1, tx1), cy1 = fmaxf(qy1, ty1);
            const float ac  = (cx1 - cx0) * (cy1 - cy0);
            // giou = iou - (ac-uni)/ac = inter/uni + uni/ac - 1
            const float giou = inter * __builtin_amdgcn_rcpf(uni) +
                               uni   * __builtin_amdgcn_rcpf(ac) - 1.0f;

            float c = 5.0f * l1 - acc[r] - 2.0f * giou;
            c = __builtin_isnan(c) ? kBIG : fminf(fmaxf(c, -kBIG), kBIG);
            out[(size_t)(r0 + m) * kN + (n0 + row)] = c;
        }
    }
}

extern "C" void kernel_launch(void* const* d_in, const int* in_sizes, int n_in,
                              void* d_out, int out_size, void* d_ws, size_t ws_size,
                              hipStream_t stream) {
    (void)in_sizes; (void)n_in; (void)out_size; (void)d_ws; (void)ws_size;
    const float* pred_logits = (const float*)d_in[0];
    const float* pred_boxes  = (const float*)d_in[1];
    const int*   tgt_ids     = (const int*)d_in[2];
    const float* tgt_boxes   = (const float*)d_in[3];
    float* out = (float*)d_out;

    dim3 grid(kBQ / kQT);   // 900 blocks, each owns a 16-row band x all targets
    dim3 block(256);        // 8 wave32 waves
    matcher_cost_kernel<<<grid, block, 0, stream>>>(pred_logits, pred_boxes,
                                                    tgt_ids, tgt_boxes, out);
}
